// TransformerBlock_20538533609811
// MI455X (gfx1250) — matrix operand order repaired and checked by reading, not compiled, not run on hardware
//
#include <hip/hip_runtime.h>

// ---------------------------------------------------------------------------
// MI455X (gfx1250) transformer block.
// Compute-bound (~73 GFLOP vs ~6.5us of HBM traffic at 23.3 TB/s), so all
// GEMM-like contractions run on v_wmma_f32_16x16x32_bf16 with fp32
// accumulation; softmax / layernorm / bias / residual stay in fp32 VALU.
// Attention K/V tiles are staged HBM->LDS with the CDNA5 async path
// (global_load_async_to_lds_b128 + s_wait_asynccnt), double-buffered so the
// next tile streams in while the current tile is consumed by WMMAs.
// ---------------------------------------------------------------------------

#define B_   4
#define S_   2048
#define D_   512
#define H_   8
#define DH_  64
#define DFF_ 2048
#define M_   (B_ * S_)          // 8192 rows
#define EPS_ 1e-5f

typedef __bf16 bf16_t;
typedef __attribute__((ext_vector_type(16))) __bf16 v16bf;
typedef __attribute__((ext_vector_type(8)))  float  v8f;

static __device__ __forceinline__ bf16_t f2bf(float f) {
  unsigned u = __builtin_bit_cast(unsigned, f);
  unsigned r = u + 0x7FFFu + ((u >> 16) & 1u);      // round-to-nearest-even
  unsigned short h = (unsigned short)(r >> 16);
  return __builtin_bit_cast(bf16_t, h);
}

static __device__ __forceinline__ v8f wmma_bf16(v16bf a, v16bf b, v8f c) {
  // (neg_a, A, neg_b, B, c_mod, C, reuse_a, reuse_b)
  return __builtin_amdgcn_wmma_f32_16x16x32_bf16(false, a, false, b,
                                                 (short)0, c, false, false);
}

// A fragment: 16x32 (MxK) row-major tile at p, leading dim ld (elements).
// Lane L holds row L%16; elems i -> K = 16*(L/16)+i.
static __device__ __forceinline__ v16bf load_a_frag(const bf16_t* p, int ld) {
  int lane = threadIdx.x & 31;
  const bf16_t* r = p + (size_t)(lane & 15) * ld + ((lane >> 4) << 4);
  v16bf a;
#pragma unroll
  for (int i = 0; i < 16; ++i) a[i] = r[i];
  return a;
}

// B fragment where B(k,n) = W[n][k]  (weight^T / K^T).  W row-major, ld = K.
// Lane L holds col n = L%16; elems i -> K = 16*(L/16)+i (contiguous in W row).
static __device__ __forceinline__ v16bf load_bT_frag(const bf16_t* w, int ld) {
  int lane = threadIdx.x & 31;
  const bf16_t* r = w + (size_t)(lane & 15) * ld + ((lane >> 4) << 4);
  v16bf b;
#pragma unroll
  for (int i = 0; i < 16; ++i) b[i] = r[i];
  return b;
}

// B fragment, row-major B[k][n] (contraction along rows), leading dim ld.
static __device__ __forceinline__ v16bf load_b_frag(const bf16_t* p, int ld) {
  int lane = threadIdx.x & 31;
  const bf16_t* r = p + (lane & 15) + (size_t)((lane >> 4) << 4) * ld;
  v16bf b;
#pragma unroll
  for (int i = 0; i < 16; ++i) b[i] = r[(size_t)i * ld];
  return b;
}

// reductions across the 16 lanes of a half-wave (same C-frag row set)
static __device__ __forceinline__ float red16_max(float v) {
#pragma unroll
  for (int m = 1; m <= 8; m <<= 1) v = fmaxf(v, __shfl_xor(v, m, 32));
  return v;
}
static __device__ __forceinline__ float red16_sum(float v) {
#pragma unroll
  for (int m = 1; m <= 8; m <<= 1) v += __shfl_xor(v, m, 32);
  return v;
}

// ---------------------------------------------------------------------------
// fp32 -> bf16 cast, 4 elements / thread
// ---------------------------------------------------------------------------
__global__ void k_f2bf4(const float* __restrict__ s, bf16_t* __restrict__ d,
                        int n) {
  int i = (blockIdx.x * blockDim.x + threadIdx.x) << 2;
  if (i >= n) return;
  float4 f = *(const float4*)(s + i);
  d[i + 0] = f2bf(f.x);
  d[i + 1] = f2bf(f.y);
  d[i + 2] = f2bf(f.z);
  d[i + 3] = f2bf(f.w);
}

// ---------------------------------------------------------------------------
// Flash attention.  grid = B*H*(S/64), block = 128 (4 waves).
// Each wave owns a 16-row Q tile; the 4 waves share one (b,h) so the 32x64
// K/V tile (K==V==x) is staged once per iteration -- asynchronously, double
// buffered, via global_load_async_to_lds_b128 (ASYNCcnt).
// ---------------------------------------------------------------------------
__global__ __launch_bounds__(128, 1) void k_attn(const bf16_t* __restrict__ xq,
                                                 bf16_t* __restrict__ ctxb) {
  __shared__ __align__(16) bf16_t ldsKV[2][32 * DH_];  // 2 x 4 KB K/V tiles
  __shared__ __align__(16) bf16_t ldsP[4][16 * 32];    // 4 KB, P staging

  const int tid  = threadIdx.x;
  const int wave = tid >> 5;
  const int lane = tid & 31;
  const int half = lane >> 4;

  const int bid = blockIdx.x;
  const int qt  = bid & 31;            // S/64 = 32 q tiles
  const int bh  = bid >> 5;
  const int h   = bh & (H_ - 1);
  const int b   = bh >> 3;

  const int    q0      = qt * 64 + wave * 16;
  const size_t rowbase = (size_t)b * S_ * D_ + (size_t)h * DH_;

  // Per-thread slice of the cooperative 32x64 tile copy: 32 bytes each.
  const int kv_st = tid >> 2;
  const int dh_st = (tid & 3) << 4;

  // Async HBM -> LDS stage of one 32x64 bf16 tile (2 x b128 per lane).
  // NOTE: INST_OFFSET is added to BOTH the LDS and the global address.
  auto stage_async = [&](int kc, int buf) {
    unsigned lds = (unsigned)(uintptr_t)(&ldsKV[buf][kv_st * DH_ + dh_st]);
    unsigned long long ga =
        (unsigned long long)(uintptr_t)(xq + rowbase +
                                        (size_t)(kc + kv_st) * D_ + dh_st);
    asm volatile(
        "global_load_async_to_lds_b128 %0, %1, off\n\t"
        "global_load_async_to_lds_b128 %0, %1, off offset:16"
        :
        : "v"(lds), "v"(ga)
        : "memory");
  };

  // Q fragments for dh = 0..31 and 32..63 (contraction dim of Q.K^T)
  v16bf qa0 = load_a_frag(xq + rowbase + (size_t)q0 * D_, D_);
  v16bf qa1 = load_a_frag(xq + rowbase + (size_t)q0 * D_ + 32, D_);

  float mrow[8], lrow[8];
  v8f   o[4];
#pragma unroll
  for (int j = 0; j < 8; ++j) { mrow[j] = -1e30f; lrow[j] = 0.f; }
#pragma unroll
  for (int t = 0; t < 4; ++t) { v8f z = {}; o[t] = z; }

  stage_async(0, 0);                       // prologue: tile 0 in flight

  for (int it = 0; it < S_ / 32; ++it) {
    const int buf = it & 1;
    // my slice of tile `it` has landed ...
    asm volatile("s_wait_asynccnt 0x0" ::: "memory");
    // ... and after the barrier every wave's slice has landed.
    __syncthreads();
    // stream the next tile into the other buffer while we compute
    if (it + 1 < S_ / 32) stage_async((it + 1) * 32, buf ^ 1);

    const bf16_t* KV = ldsKV[buf];

    // S = (Q/sqrt(DH)) . K^T :  16 x 32 scores = two C frags, K=64 = 2 steps
    v8f s0 = {}, s1 = {};
    s0 = wmma_bf16(qa0, load_bT_frag(KV, DH_), s0);
    s0 = wmma_bf16(qa1, load_bT_frag(KV + 32, DH_), s0);
    s1 = wmma_bf16(qa0, load_bT_frag(KV + 16 * DH_, DH_), s1);
    s1 = wmma_bf16(qa1, load_bT_frag(KV + 16 * DH_ + 32, DH_), s1);

    // online softmax (rows j + 8*half, cols across 16 lanes of the half)
    float alpha[8];
#pragma unroll
    for (int j = 0; j < 8; ++j) {
      float a0 = s0[j] * 0.125f;           // 1/sqrt(64)
      float a1 = s1[j] * 0.125f;
      s0[j] = a0; s1[j] = a1;
      float mnew = fmaxf(mrow[j], red16_max(fmaxf(a0, a1)));
      alpha[j] = __expf(mrow[j] - mnew);
      mrow[j]  = mnew;
    }
#pragma unroll
    for (int j = 0; j < 8; ++j) {
      float p0 = __expf(s0[j] - mrow[j]);
      float p1 = __expf(s1[j] - mrow[j]);
      lrow[j] = lrow[j] * alpha[j] + red16_sum(p0 + p1);
      int r = (j + (half << 3)) * 32;
      ldsP[wave][r + (lane & 15)]      = f2bf(p0);  // D-frag -> A-frag via LDS
      ldsP[wave][r + 16 + (lane & 15)] = f2bf(p1);
#pragma unroll
      for (int t = 0; t < 4; ++t) o[t][j] *= alpha[j];
    }
    __syncthreads();

    // O += P(16x32) . V(32x64)
    v16bf pa = load_a_frag(ldsP[wave], 32);
#pragma unroll
    for (int t = 0; t < 4; ++t)
      o[t] = wmma_bf16(pa, load_b_frag(KV + 16 * t, DH_), o[t]);
  }

  // normalize and emit ctx (heads merged: d = h*64 + dh) as bf16
#pragma unroll
  for (int j = 0; j < 8; ++j) {
    float inv = 1.f / lrow[j];
    int   srow = q0 + j + (half << 3);
    size_t base = (size_t)b * S_ * D_ + (size_t)srow * D_ + h * DH_ + (lane & 15);
#pragma unroll
    for (int t = 0; t < 4; ++t) ctxb[base + 16 * t] = f2bf(o[t][j] * inv);
  }
}

// ---------------------------------------------------------------------------
// out = LayerNorm(A[M,K] @ W[N,K]^T + bias) * lns + lnb + resid, N = D_ = 512.
// One block (8 waves) owns 16 full rows so LN needs only an LDS reduction.
// Used for both the Wo path (K=512) and the FFN2 path (K=2048).
// ---------------------------------------------------------------------------
__global__ __launch_bounds__(256, 1) void k_gemm_ln(
    const bf16_t* __restrict__ A, const bf16_t* __restrict__ W,
    const float* __restrict__ bias, const float* __restrict__ lns,
    const float* __restrict__ lnb, const float* __restrict__ resid,
    float* __restrict__ outf, bf16_t* __restrict__ outb, int K) {
  __shared__ float redS[16][8];
  __shared__ float redQ[16][8];

  const int wave = threadIdx.x >> 5;
  const int lane = threadIdx.x & 31;
  const int half = lane >> 4;
  const int m0   = blockIdx.x * 16;
  const int nb   = wave * 64;

  v8f acc[4];
#pragma unroll
  for (int t = 0; t < 4; ++t) { v8f z = {}; acc[t] = z; }

  for (int k0 = 0; k0 < K; k0 += 32) {
    __builtin_prefetch(A + (size_t)m0 * K + k0 + 64, 0, 1);
    v16bf a = load_a_frag(A + (size_t)m0 * K + k0, K);
#pragma unroll
    for (int t = 0; t < 4; ++t) {
      v16bf bf = load_bT_frag(W + (size_t)(nb + 16 * t) * K + k0, K);
      acc[t] = wmma_bf16(a, bf, acc[t]);
    }
  }

  // bias + per-row partial sums for LN
  float ps[8], pq[8];
#pragma unroll
  for (int j = 0; j < 8; ++j) { ps[j] = 0.f; pq[j] = 0.f; }
#pragma unroll
  for (int t = 0; t < 4; ++t) {
    float bv = bias[nb + 16 * t + (lane & 15)];
#pragma unroll
    for (int j = 0; j < 8; ++j) {
      float v = acc[t][j] + bv;
      acc[t][j] = v;
      ps[j] += v;
      pq[j] += v * v;
    }
  }
#pragma unroll
  for (int j = 0; j < 8; ++j) {
    float s = red16_sum(ps[j]);
    float q = red16_sum(pq[j]);
    if ((lane & 15) == 0) {
      redS[j + half * 8][wave] = s;
      redQ[j + half * 8][wave] = q;
    }
  }
  __syncthreads();

#pragma unroll
  for (int j = 0; j < 8; ++j) {
    int r = j + half * 8;
    float s = 0.f, q = 0.f;
#pragma unroll
    for (int w = 0; w < 8; ++w) { s += redS[r][w]; q += redQ[r][w]; }
    float mean = s * (1.f / D_);
    float var  = q * (1.f / D_) - mean * mean;
    float rstd = rsqrtf(var + EPS_);
    size_t row = (size_t)(m0 + r) * D_;
#pragma unroll
    for (int t = 0; t < 4; ++t) {
      int n = nb + 16 * t + (lane & 15);
      float v = (acc[t][j] - mean) * rstd * lns[n] + lnb[n] + resid[row + n];
      outf[row + n] = v;
      if (outb) outb[row + n] = f2bf(v);
    }
  }
}

// ---------------------------------------------------------------------------
// h = relu(A[M,512] @ W1[N,512]^T + b1) -> bf16.  Block tile 128x128, 8 waves.
// ---------------------------------------------------------------------------
__global__ __launch_bounds__(256, 1) void k_gemm_relu(
    const bf16_t* __restrict__ A, const bf16_t* __restrict__ W,
    const float* __restrict__ bias, bf16_t* __restrict__ out) {
  const int wave = threadIdx.x >> 5;
  const int lane = threadIdx.x & 31;
  const int half = lane >> 4;
  const int m0   = blockIdx.x * 128 + wave * 16;
  const int n0   = blockIdx.y * 128;

  v8f acc[8];
#pragma unroll
  for (int t = 0; t < 8; ++t) { v8f z = {}; acc[t] = z; }

  for (int k0 = 0; k0 < D_; k0 += 32) {
    __builtin_prefetch(W + (size_t)n0 * D_ + k0 + 64, 0, 1);
    v16bf a = load_a_frag(A + (size_t)m0 * D_ + k0, D_);
#pragma unroll
    for (int t = 0; t < 8; ++t) {
      v16bf bf = load_bT_frag(W + (size_t)(n0 + 16 * t) * D_ + k0, D_);
      acc[t] = wmma_bf16(a, bf, acc[t]);
    }
  }
#pragma unroll
  for (int t = 0; t < 8; ++t) {
    int n = n0 + 16 * t + (lane & 15);
    float bv = bias[n];
#pragma unroll
    for (int j = 0; j < 8; ++j) {
      float v = fmaxf(acc[t][j] + bv, 0.f);
      out[(size_t)(m0 + j + half * 8) * DFF_ + n] = f2bf(v);
    }
  }
}

// ---------------------------------------------------------------------------
extern "C" void kernel_launch(void* const* d_in, const int* in_sizes, int n_in,
                              void* d_out, int out_size, void* d_ws,
                              size_t ws_size, hipStream_t stream) {
  const float* x     = (const float*)d_in[0];
  // d_in[1] = p, d_in[2] = e : unused by the reference forward pass
  const float* Wo_w  = (const float*)d_in[3];
  const float* Wo_b  = (const float*)d_in[4];
  const float* ln_as = (const float*)d_in[5];
  const float* ln_ab = (const float*)d_in[6];
  const float* W1_w  = (const float*)d_in[7];
  const float* W1_b  = (const float*)d_in[8];
  const float* W2_w  = (const float*)d_in[9];
  const float* W2_b  = (const float*)d_in[10];
  const float* ln_fs = (const float*)d_in[11];
  const float* ln_fb = (const float*)d_in[12];

  char*  ws  = (char*)d_ws;
  size_t off = 0;
  auto alloc = [&](size_t bytes) -> void* {
    void* p = ws + off;
    off += (bytes + 255) & ~(size_t)255;
    return p;
  };
  bf16_t* xb   = (bf16_t*)alloc((size_t)M_ * D_ * 2);     // x in bf16
  bf16_t* ctxb = (bf16_t*)alloc((size_t)M_ * D_ * 2);     // attention context
  bf16_t* x1b  = (bf16_t*)alloc((size_t)M_ * D_ * 2);     // x1 in bf16
  bf16_t* hb   = (bf16_t*)alloc((size_t)M_ * DFF_ * 2);   // relu(ffn1)
  bf16_t* Wob  = (bf16_t*)alloc((size_t)D_ * D_ * 2);
  bf16_t* W1b  = (bf16_t*)alloc((size_t)DFF_ * D_ * 2);
  bf16_t* W2b  = (bf16_t*)alloc((size_t)D_ * DFF_ * 2);
  float*  x1f  = (float*)alloc((size_t)M_ * D_ * 4);      // x1 in fp32

  // 1) cast activations + weights to bf16
  k_f2bf4<<<(M_ * D_) / 1024, 256, 0, stream>>>(x, xb, M_ * D_);
  k_f2bf4<<<(D_ * D_) / 1024, 256, 0, stream>>>(Wo_w, Wob, D_ * D_);
  k_f2bf4<<<(DFF_ * D_) / 1024, 256, 0, stream>>>(W1_w, W1b, DFF_ * D_);
  k_f2bf4<<<(D_ * DFF_) / 1024, 256, 0, stream>>>(W2_w, W2b, D_ * DFF_);

  // 2) flash attention (q=k=v=x), heads merged on output
  k_attn<<<B_ * H_ * (S_ / 64), 128, 0, stream>>>(xb, ctxb);

  // 3) x1 = LN(ctx @ Wo^T + Wo_b) + x
  k_gemm_ln<<<M_ / 16, 256, 0, stream>>>(ctxb, Wob, Wo_b, ln_as, ln_ab, x, x1f,
                                         x1b, D_);

  // 4) h = relu(x1 @ W1^T + W1_b)
  k_gemm_relu<<<dim3(M_ / 128, DFF_ / 128), 256, 0, stream>>>(x1b, W1b, W1_b,
                                                              hb);

  // 5) out = LN(h @ W2^T + W2_b) + x1
  k_gemm_ln<<<M_ / 16, 256, 0, stream>>>(hb, W2b, W2_b, ln_fs, ln_fb, x1f,
                                         (float*)d_out, (bf16_t*)nullptr, DFF_);
}
